// QuantumAttentionBlock_50122268344526
// MI455X (gfx1250) — matrix operand order repaired and checked by reading, not compiled, not run on hardware
//
#include <hip/hip_runtime.h>
#include <math.h>

typedef __attribute__((ext_vector_type(16))) _Float16 v16h;
typedef __attribute__((ext_vector_type(8)))  _Float16 v8h;
typedef __attribute__((ext_vector_type(8)))  float    v8f;

#define NB    4           // batches
#define NSEQ  2048
#define DM    1024
#define MTOT  (NB * NSEQ) // 8192

// ---------------------------------------------------------------------------
// WMMA fragment loaders, per CDNA5 ISA 7.12.2 (wave32, 16-bit operands)
// A 16x32 (MxK): lane L holds row M=L&15. halves 0..7 -> K = kb+0..7,
// halves 8..15 -> K = 16+kb+0..7 where kb = (L<16)?0:8. Two 16B loads.
// ---------------------------------------------------------------------------
__device__ __forceinline__ v16h load_frag_a(const _Float16* __restrict__ p, int ld, int lane) {
  const _Float16* q = p + (size_t)(lane & 15) * ld + ((lane < 16) ? 0 : 8);
  union { v16h v; v8h h[2]; } u;
  u.h[0] = *(const v8h*)(q);
  u.h[1] = *(const v8h*)(q + 16);
  return u.v;
}
// B 32x16 (KxN) with B[k,n] = Bt[n,k] (Bt row-major, ld over k): lane L holds
// col N=L&15; halves 0..7 -> K = kb+0..7, halves 8..15 -> K = 16+kb+0..7, kb = (L<16)?0:8. Two 16B loads.
__device__ __forceinline__ v16h load_frag_b(const _Float16* __restrict__ p, int ld, int lane) {
  const _Float16* q = p + (size_t)(lane & 15) * ld + ((lane < 16) ? 0 : 8);
  union { v16h v; v8h h[2]; } u;
  u.h[0] = *(const v8h*)(q);
  u.h[1] = *(const v8h*)(q + 16);
  return u.v;
}

#define WMMA_F16(a, b, c) \
  __builtin_amdgcn_wmma_f32_16x16x32_f16(false, (a), false, (b), (short)0, (c), false, false)

// ---------------------------------------------------------------------------
// fp32 -> f16 convert
// ---------------------------------------------------------------------------
__global__ void cvt_f32_f16_kernel(const float* __restrict__ s, _Float16* __restrict__ d, size_t n) {
  size_t i = (size_t)blockIdx.x * blockDim.x + threadIdx.x;
  if (i < n) d[i] = (_Float16)s[i];
}

// ---------------------------------------------------------------------------
// Generic batched WMMA GEMM: Y[b][m][n] = sum_k A[b][m][k] * Bt[b][n][k] + bias[n]
// One wave per 32x64 output tile: 2 A frags + 4 B frags -> 8 WMMAs per k-step.
// transStore=1 writes Y transposed per batch: Y[b][n][m] (used for V^T).
// ---------------------------------------------------------------------------
__global__ void wmma_gemm_f16_kernel(const _Float16* __restrict__ A, size_t sA,
                                     const _Float16* __restrict__ Bt, size_t sB,
                                     const float* __restrict__ bias,
                                     _Float16* __restrict__ Y, size_t sY,
                                     int M, int K, int N, int transStore) {
  const int lane = threadIdx.x;
  const _Float16* Ab = A + (size_t)blockIdx.z * sA;
  const _Float16* Bb = Bt + (size_t)blockIdx.z * sB;
  _Float16* Yb = Y + (size_t)blockIdx.z * sY;
  const int m0 = blockIdx.x * 32;
  const int n0 = blockIdx.y * 64;

  v8f acc[2][4];
  #pragma unroll
  for (int mi = 0; mi < 2; ++mi)
    #pragma unroll
    for (int ni = 0; ni < 4; ++ni) acc[mi][ni] = (v8f){};

  const _Float16* Ap0 = Ab + (size_t)(m0 +  0) * K;
  const _Float16* Ap1 = Ab + (size_t)(m0 + 16) * K;
  const _Float16* Bp0 = Bb + (size_t)(n0 +  0) * K;
  const _Float16* Bp1 = Bb + (size_t)(n0 + 16) * K;
  const _Float16* Bp2 = Bb + (size_t)(n0 + 32) * K;
  const _Float16* Bp3 = Bb + (size_t)(n0 + 48) * K;

  for (int k0 = 0; k0 < K; k0 += 32) {
    v16h a0 = load_frag_a(Ap0 + k0, K, lane);
    v16h a1 = load_frag_a(Ap1 + k0, K, lane);
    v16h b0 = load_frag_b(Bp0 + k0, K, lane);
    v16h b1 = load_frag_b(Bp1 + k0, K, lane);
    v16h b2 = load_frag_b(Bp2 + k0, K, lane);
    v16h b3 = load_frag_b(Bp3 + k0, K, lane);
    acc[0][0] = WMMA_F16(a0, b0, acc[0][0]);
    acc[0][1] = WMMA_F16(a0, b1, acc[0][1]);
    acc[0][2] = WMMA_F16(a0, b2, acc[0][2]);
    acc[0][3] = WMMA_F16(a0, b3, acc[0][3]);
    acc[1][0] = WMMA_F16(a1, b0, acc[1][0]);
    acc[1][1] = WMMA_F16(a1, b1, acc[1][1]);
    acc[1][2] = WMMA_F16(a1, b2, acc[1][2]);
    acc[1][3] = WMMA_F16(a1, b3, acc[1][3]);
  }

  const int nl = lane & 15;
  const int moff = (lane < 16) ? 0 : 8;
  #pragma unroll
  for (int mi = 0; mi < 2; ++mi) {
    const int mb = m0 + mi * 16 + moff;
    #pragma unroll
    for (int t = 0; t < 4; ++t) {
      int n = n0 + t * 16 + nl;
      float bv = bias ? bias[n] : 0.0f;
      #pragma unroll
      for (int r = 0; r < 8; ++r) {
        float v = acc[mi][t][r] + bv;
        if (!transStore) Yb[(size_t)(mb + r) * N + n] = (_Float16)v;
        else             Yb[(size_t)n * M + (mb + r)] = (_Float16)v;
      }
    }
  }
}

// ---------------------------------------------------------------------------
// qp = X @ Wtq^T + btq : [MTOT, 2]. One wave per row, fp32.
// ---------------------------------------------------------------------------
__global__ void qp_kernel(const float* __restrict__ X, const float* __restrict__ Wtq,
                          const float* __restrict__ btq, float* __restrict__ qp) {
  const int wave = threadIdx.x >> 5;
  const int lane = threadIdx.x & 31;
  const int row = blockIdx.x * 4 + wave;
  const float* x = X + (size_t)row * DM;
  float s0 = 0.f, s1 = 0.f;
  for (int k = lane; k < DM; k += 32) {
    float xv = x[k];
    s0 += xv * Wtq[k];
    s1 += xv * Wtq[DM + k];
  }
  for (int off = 16; off > 0; off >>= 1) {
    s0 += __shfl_xor(s0, off, 32);
    s1 += __shfl_xor(s1, off, 32);
  }
  if (lane == 0) {
    qp[(size_t)row * 2 + 0] = s0 + btq[0];
    qp[(size_t)row * 2 + 1] = s1 + btq[1];
  }
}

// ---------------------------------------------------------------------------
// Per-batch sum(dist^2), sum(dist^4) over all NxN pairs.
// ---------------------------------------------------------------------------
__global__ void zero_stats_kernel(float* stats) {
  if (threadIdx.x < 8) stats[threadIdx.x] = 0.0f;
}

__global__ void stats_kernel(const float* __restrict__ qp, float* __restrict__ stats) {
  __shared__ float rs[256], rss[256];
  const int b = blockIdx.y;
  const float* q = qp + (size_t)b * NSEQ * 2;
  const int tid = threadIdx.x;
  const size_t start = (size_t)blockIdx.x * 16384;
  float s = 0.f, ss = 0.f;
  for (int it = 0; it < 64; ++it) {
    size_t idx = start + (size_t)it * 256 + tid;
    int i = (int)(idx >> 11);
    int j = (int)(idx & (NSEQ - 1));
    float dx = q[i * 2 + 0] - q[j * 2 + 0];
    float dy = q[i * 2 + 1] - q[j * 2 + 1];
    float d2 = dx * dx + dy * dy;
    s += d2; ss += d2 * d2;
  }
  rs[tid] = s; rss[tid] = ss;
  __syncthreads();
  for (int o = 128; o > 0; o >>= 1) {
    if (tid < o) { rs[tid] += rs[tid + o]; rss[tid] += rss[tid + o]; }
    __syncthreads();
  }
  if (tid == 0) {
    atomicAdd(&stats[b * 2 + 0], rs[0]);
    atomicAdd(&stats[b * 2 + 1], rss[0]);
  }
}

// ---------------------------------------------------------------------------
// scores = (Q K^T)/sqrt(D), fused phase/threshold/causal mask -> fp32 S.
// One wave per 16x64 strip: A fragment reused across 4 j-tiles.
// cos(atan2(dy,dx)) == dx/sqrt(dx^2+dy^2); bias only gates via ==0 test.
// ---------------------------------------------------------------------------
__global__ void scores_kernel(const _Float16* __restrict__ Q, const _Float16* __restrict__ Kh,
                              const float* __restrict__ qp, const float* __restrict__ stats,
                              float* __restrict__ S) {
  const int lane = threadIdx.x;
  const int b = blockIdx.z;
  const _Float16* Qb = Q + (size_t)b * NSEQ * DM;
  const _Float16* Kb = Kh + (size_t)b * NSEQ * DM;
  const int m0 = blockIdx.y * 16;
  const int n0 = blockIdx.x * 64;

  v8f acc[4];
  #pragma unroll
  for (int t = 0; t < 4; ++t) acc[t] = (v8f){};

  const _Float16* Ap  = Qb + (size_t)m0 * DM;
  const _Float16* Bp0 = Kb + (size_t)(n0 +  0) * DM;
  const _Float16* Bp1 = Kb + (size_t)(n0 + 16) * DM;
  const _Float16* Bp2 = Kb + (size_t)(n0 + 32) * DM;
  const _Float16* Bp3 = Kb + (size_t)(n0 + 48) * DM;
  for (int k0 = 0; k0 < DM; k0 += 32) {
    v16h a  = load_frag_a(Ap  + k0, DM, lane);
    v16h b0 = load_frag_b(Bp0 + k0, DM, lane);
    v16h b1 = load_frag_b(Bp1 + k0, DM, lane);
    v16h b2 = load_frag_b(Bp2 + k0, DM, lane);
    v16h b3 = load_frag_b(Bp3 + k0, DM, lane);
    acc[0] = WMMA_F16(a, b0, acc[0]);
    acc[1] = WMMA_F16(a, b1, acc[1]);
    acc[2] = WMMA_F16(a, b2, acc[2]);
    acc[3] = WMMA_F16(a, b3, acc[3]);
  }

  const float cnt = (float)NSEQ * (float)NSEQ;
  const float sum = stats[b * 2 + 0], ssq = stats[b * 2 + 1];
  const float mean = sum / cnt;
  const float var = (ssq - sum * sum / cnt) / (cnt - 1.0f);   // unbiased
  const float thr = mean + 1.25f * sqrtf(fmaxf(var, 0.0f));

  const float* qb = qp + (size_t)b * NSEQ * 2;
  const int mb = m0 + ((lane < 16) ? 0 : 8);
  float qix[8], qiy[8];
  #pragma unroll
  for (int r = 0; r < 8; ++r) { qix[r] = qb[(mb + r) * 2 + 0]; qiy[r] = qb[(mb + r) * 2 + 1]; }

  float* Sb = S + (size_t)b * NSEQ * NSEQ;
  const float scale = 0.03125f;  // 1/sqrt(1024)
  #pragma unroll
  for (int t = 0; t < 4; ++t) {
    const int j = n0 + t * 16 + (lane & 15);
    const float qjx = qb[j * 2 + 0], qjy = qb[j * 2 + 1];
    #pragma unroll
    for (int r = 0; r < 8; ++r) {
      int i = mb + r;
      float dx = qix[r] - qjx;
      float dy = qiy[r] - qjy;
      float d2 = dx * dx + dy * dy;
      float fb = (d2 > 0.0f) ? 0.5f * (1.0f + dx / sqrtf(d2)) : 1.0f;
      bool keep = (j <= i) && (d2 <= thr) && (fb != 0.0f);
      Sb[(size_t)i * NSEQ + j] = keep ? acc[t][r] * scale : -__builtin_inff();
    }
  }
}

// ---------------------------------------------------------------------------
// Row softmax over N=2048, output f16 probabilities.
// ---------------------------------------------------------------------------
__global__ void softmax_kernel(const float* __restrict__ S, _Float16* __restrict__ P) {
  __shared__ float red[256];
  const size_t row = blockIdx.x;
  const float* s = S + row * NSEQ;
  _Float16* p = P + row * NSEQ;
  const int tid = threadIdx.x;
  float vals[8];
  float mx = -__builtin_inff();
  #pragma unroll
  for (int t = 0; t < 8; ++t) { vals[t] = s[tid + t * 256]; mx = fmaxf(mx, vals[t]); }
  red[tid] = mx; __syncthreads();
  for (int o = 128; o > 0; o >>= 1) { if (tid < o) red[tid] = fmaxf(red[tid], red[tid + o]); __syncthreads(); }
  mx = red[0]; __syncthreads();
  float sum = 0.f;
  #pragma unroll
  for (int t = 0; t < 8; ++t) { vals[t] = expf(vals[t] - mx); sum += vals[t]; }
  red[tid] = sum; __syncthreads();
  for (int o = 128; o > 0; o >>= 1) { if (tid < o) red[tid] += red[tid + o]; __syncthreads(); }
  const float inv = 1.0f / red[0];
  #pragma unroll
  for (int t = 0; t < 8; ++t) p[tid + t * 256] = (_Float16)(vals[t] * inv);
}

// ---------------------------------------------------------------------------
// out = hidden + (ctx @ Wo^T + bo) + 0.1*(normalize(qp) @ Wfq^T + bfq)
// One wave per 32x64 tile, fp32 epilogue with residual + quantum term.
// ---------------------------------------------------------------------------
__global__ void out_gemm_kernel(const _Float16* __restrict__ Ctx, const _Float16* __restrict__ Wo,
                                const float* __restrict__ bo, const float* __restrict__ hidden,
                                const float* __restrict__ qp, const float* __restrict__ Wfq,
                                const float* __restrict__ bfq, float* __restrict__ out) {
  const int lane = threadIdx.x;
  const int m0 = blockIdx.x * 32;
  const int n0 = blockIdx.y * 64;

  v8f acc[2][4];
  #pragma unroll
  for (int mi = 0; mi < 2; ++mi)
    #pragma unroll
    for (int ni = 0; ni < 4; ++ni) acc[mi][ni] = (v8f){};

  const _Float16* Ap0 = Ctx + (size_t)(m0 +  0) * DM;
  const _Float16* Ap1 = Ctx + (size_t)(m0 + 16) * DM;
  const _Float16* Bp0 = Wo + (size_t)(n0 +  0) * DM;
  const _Float16* Bp1 = Wo + (size_t)(n0 + 16) * DM;
  const _Float16* Bp2 = Wo + (size_t)(n0 + 32) * DM;
  const _Float16* Bp3 = Wo + (size_t)(n0 + 48) * DM;
  for (int k0 = 0; k0 < DM; k0 += 32) {
    v16h a0 = load_frag_a(Ap0 + k0, DM, lane);
    v16h a1 = load_frag_a(Ap1 + k0, DM, lane);
    v16h b0 = load_frag_b(Bp0 + k0, DM, lane);
    v16h b1 = load_frag_b(Bp1 + k0, DM, lane);
    v16h b2 = load_frag_b(Bp2 + k0, DM, lane);
    v16h b3 = load_frag_b(Bp3 + k0, DM, lane);
    acc[0][0] = WMMA_F16(a0, b0, acc[0][0]);
    acc[0][1] = WMMA_F16(a0, b1, acc[0][1]);
    acc[0][2] = WMMA_F16(a0, b2, acc[0][2]);
    acc[0][3] = WMMA_F16(a0, b3, acc[0][3]);
    acc[1][0] = WMMA_F16(a1, b0, acc[1][0]);
    acc[1][1] = WMMA_F16(a1, b1, acc[1][1]);
    acc[1][2] = WMMA_F16(a1, b2, acc[1][2]);
    acc[1][3] = WMMA_F16(a1, b3, acc[1][3]);
  }

  const int nl = lane & 15;
  const int moff = (lane < 16) ? 0 : 8;
  #pragma unroll
  for (int mi = 0; mi < 2; ++mi) {
    const int mb = m0 + mi * 16 + moff;
    // per-row quantum epilogue terms (shared across the 4 n-tiles)
    float qnx[8], qny[8];
    #pragma unroll
    for (int r = 0; r < 8; ++r) {
      float qx = qp[(mb + r) * 2 + 0], qy = qp[(mb + r) * 2 + 1];
      float inv = 1.0f / fmaxf(sqrtf(qx * qx + qy * qy), 1e-12f);
      qnx[r] = qx * inv; qny[r] = qy * inv;
    }
    #pragma unroll
    for (int t = 0; t < 4; ++t) {
      const int n = n0 + t * 16 + nl;
      const float bov = bo[n];
      const float wf0 = Wfq[n * 2 + 0], wf1 = Wfq[n * 2 + 1], bfv = bfq[n];
      #pragma unroll
      for (int r = 0; r < 8; ++r) {
        int m = mb + r;
        float qo = qnx[r] * wf0 + qny[r] * wf1 + bfv;
        out[(size_t)m * DM + n] =
            hidden[(size_t)m * DM + n] + acc[mi][t][r] + bov + 0.1f * qo;
      }
    }
  }
}

// ---------------------------------------------------------------------------
// Launcher
// ---------------------------------------------------------------------------
extern "C" void kernel_launch(void* const* d_in, const int* in_sizes, int n_in,
                              void* d_out, int out_size, void* d_ws, size_t ws_size,
                              hipStream_t stream) {
  (void)in_sizes; (void)n_in; (void)out_size; (void)ws_size;
  const float* hidden = (const float*)d_in[0];
  const float* Wq = (const float*)d_in[1];  const float* bq = (const float*)d_in[2];
  const float* Wk = (const float*)d_in[3];  const float* bk = (const float*)d_in[4];
  const float* Wv = (const float*)d_in[5];  const float* bv = (const float*)d_in[6];
  const float* Wo = (const float*)d_in[7];  const float* bo = (const float*)d_in[8];
  const float* Wtq = (const float*)d_in[9]; const float* btq = (const float*)d_in[10];
  const float* Wfq = (const float*)d_in[11];const float* bfq = (const float*)d_in[12];
  float* out = (float*)d_out;

  // workspace layout (256B aligned slices)
  char* w = (char*)d_ws;
  auto alloc = [&](size_t bytes) { void* p = (void*)w; w += (bytes + 255) & ~(size_t)255; return p; };
  const size_t nX = (size_t)MTOT * DM;          // 8M
  const size_t nW = (size_t)DM * DM;            // 1M
  const size_t nS = (size_t)NB * NSEQ * NSEQ;   // 16M

  _Float16* Xh  = (_Float16*)alloc(nX * 2);
  _Float16* Wqh = (_Float16*)alloc(nW * 2);
  _Float16* Wkh = (_Float16*)alloc(nW * 2);
  _Float16* Wvh = (_Float16*)alloc(nW * 2);
  _Float16* Woh = (_Float16*)alloc(nW * 2);
  _Float16* Qh  = (_Float16*)alloc(nX * 2);
  _Float16* Kh  = (_Float16*)alloc(nX * 2);
  _Float16* Vt  = (_Float16*)alloc(nX * 2);     // [b][d][j]
  _Float16* Ctx = (_Float16*)alloc(nX * 2);
  float*    qpb = (float*)alloc((size_t)MTOT * 2 * 4);
  float*    st  = (float*)alloc(256);
  float*    Sb  = (float*)alloc(nS * 4);
  _Float16* Pb  = (_Float16*)alloc(nS * 2);

  // 1) fp32 -> f16 conversions
  cvt_f32_f16_kernel<<<dim3((unsigned)((nX + 255) / 256)), 256, 0, stream>>>(hidden, Xh, nX);
  cvt_f32_f16_kernel<<<dim3((unsigned)((nW + 255) / 256)), 256, 0, stream>>>(Wq, Wqh, nW);
  cvt_f32_f16_kernel<<<dim3((unsigned)((nW + 255) / 256)), 256, 0, stream>>>(Wk, Wkh, nW);
  cvt_f32_f16_kernel<<<dim3((unsigned)((nW + 255) / 256)), 256, 0, stream>>>(Wv, Wvh, nW);
  cvt_f32_f16_kernel<<<dim3((unsigned)((nW + 255) / 256)), 256, 0, stream>>>(Wo, Woh, nW);

  // 2) Q/K projections (flat M=8192), V projection stored transposed per batch
  wmma_gemm_f16_kernel<<<dim3(MTOT / 32, DM / 64, 1), 32, 0, stream>>>(
      Xh, 0, Wqh, 0, bq, Qh, 0, MTOT, DM, DM, 0);
  wmma_gemm_f16_kernel<<<dim3(MTOT / 32, DM / 64, 1), 32, 0, stream>>>(
      Xh, 0, Wkh, 0, bk, Kh, 0, MTOT, DM, DM, 0);
  wmma_gemm_f16_kernel<<<dim3(NSEQ / 32, DM / 64, NB), 32, 0, stream>>>(
      Xh, (size_t)NSEQ * DM, Wvh, 0, bv, Vt, (size_t)DM * NSEQ, NSEQ, DM, DM, 1);

  // 3) 2-d quantum projection (fp32)
  qp_kernel<<<dim3(MTOT / 4), 128, 0, stream>>>(hidden, Wtq, btq, qpb);

  // 4) per-batch dist^2 stats
  zero_stats_kernel<<<1, 32, 0, stream>>>(st);
  stats_kernel<<<dim3(256, NB), 256, 0, stream>>>(qpb, st);

  // 5) masked scores (WMMA + fused mask), 16x64 strips
  scores_kernel<<<dim3(NSEQ / 64, NSEQ / 16, NB), 32, 0, stream>>>(Qh, Kh, qpb, st, Sb);

  // 6) softmax -> f16 probabilities
  softmax_kernel<<<dim3(NB * NSEQ), 256, 0, stream>>>(Sb, Pb);

  // 7) context = P @ V  (B-fragments read V^T rows contiguously)
  wmma_gemm_f16_kernel<<<dim3(NSEQ / 32, DM / 64, NB), 32, 0, stream>>>(
      Pb, (size_t)NSEQ * NSEQ, Vt, (size_t)DM * NSEQ, nullptr,
      Ctx, (size_t)NSEQ * DM, NSEQ, NSEQ, DM, 0);

  // 8) output projection + residual + quantum epilogue -> d_out (fp32)
  out_gemm_kernel<<<dim3(MTOT / 32, DM / 64), 32, 0, stream>>>(
      Ctx, Woh, bo, hidden, qpb, Wfq, bfq, out);
}
